// Net_24781961297991
// MI455X (gfx1250) — compile-verified, matrix operand order checked by reading, not run on hardware
//
#include <hip/hip_runtime.h>
#include <math.h>

#define N_STU  20000
#define N_EXER 10000
#define KDIM   128
#define NNODE  (N_STU + N_EXER + KDIM)   // 30128 = 16 * 1883
#define EEDGE  480000
#define BB     8192
#define BCV    2048
#define INV_TEMP 2.0f                    // 1 / 0.5

typedef __attribute__((ext_vector_type(16))) _Float16 v16h;
typedef __attribute__((ext_vector_type(8)))  float    v8f;

// ---------- helpers ----------
__device__ __forceinline__ float atomAddF(float* p, float v) {
  return unsafeAtomicAdd(p, v);        // lowers to global_atomic_add_f32
}
__device__ __forceinline__ unsigned fenc(float f) {        // monotonic float->uint
  int b = __float_as_int(f);
  return (unsigned)b ^ (unsigned)((b >> 31) | 0x80000000);
}
__device__ __forceinline__ float fdec(unsigned u) {
  int b = (u & 0x80000000u) ? (int)(u ^ 0x80000000u) : (int)(~u);
  return __int_as_float(b);
}
__device__ __forceinline__ float sigmoidf(float x) { return 1.f / (1.f + expf(-x)); }

// LDS byte offset of a shared-memory pointer (low 32 bits of LDS aperture addr)
__device__ __forceinline__ unsigned ldsOffOf(const void* p) {
  return (unsigned)(size_t)p;
}
// async copy: 32 lanes x 8B = one 256B row  (LDS offset in one VGPR, global address in a VGPR pair)
__device__ __forceinline__ void asyncLoadLdsB64(unsigned ldsByte, const void* gsrc) {
  asm volatile("global_load_async_to_lds_b64 %0, %1, off"
               :: "v"(ldsByte), "v"((unsigned long long)(size_t)gsrc)
               : "memory");
}
__device__ __forceinline__ void waitAsync0() {
  asm volatile("s_wait_asynccnt 0x0" ::: "memory");
}

// ---------- transpose + f32 -> f16 convert (for 128x128 weights) ----------
__global__ void k_transpose_f16(const float* __restrict__ W, _Float16* __restrict__ Wt) {
  int i = blockIdx.x * 256 + threadIdx.x;      // i over 128*128
  if (i >= KDIM * KDIM) return;
  int n = i >> 7, k = i & 127;
  Wt[n * KDIM + k] = (_Float16)W[k * KDIM + n];
}

// ---------- h0 = concat(stu, exer, k)  (f32 + f16 shadow) ----------
__global__ void k_build_h0(float* __restrict__ h, _Float16* __restrict__ h16,
                           const float* __restrict__ stu, const float* __restrict__ exer,
                           const float* __restrict__ ktab) {
  size_t i = (size_t)blockIdx.x * blockDim.x + threadIdx.x;
  const size_t NK = (size_t)NNODE * KDIM;
  if (i >= NK) return;
  size_t n = i >> 7;
  float v;
  if (n < N_STU)               v = stu[i];
  else if (n < N_STU + N_EXER) v = exer[i - (size_t)N_STU * KDIM];
  else                         v = ktab[i - (size_t)(N_STU + N_EXER) * KDIM];
  h[i] = v;
  h16[i] = (_Float16)v;
}

// ---------- WMMA GEMM:  C[M x Nc] = A[M x 128] @ B[Nc x 128]^T * scale ----------
// (node GEMM passes pre-transposed weights; sim GEMM passes z2 rows directly)
// grid.x = M/16 row tiles, grid.y = Nc/128 col tiles, block = 256 (8 waves).
__global__ __launch_bounds__(256)
void k_gemm128(const _Float16* __restrict__ A, const _Float16* __restrict__ B,
               float* __restrict__ C, int ldc, float scale) {
  constexpr int LD = 136;                       // padded LDS row stride (halfs)
  __shared__ _Float16 Bs[128 * LD];             // Bs[n][k]
  __shared__ _Float16 As[16 * LD];              // As[m][k]
  const int tid = threadIdx.x;
  const int colBase = blockIdx.y * 128;
  const int rowBase = blockIdx.x * 16;
  const int lane = tid & 31, wave = tid >> 5;

  // ---- async global->LDS tile copies (one 256B row per issue) ----
  for (int r = wave; r < 128; r += 8)
    asyncLoadLdsB64(ldsOffOf(&Bs[r * LD]) + lane * 8,
                    B + (size_t)(colBase + r) * 128 + lane * 4);
  for (int r = wave; r < 16; r += 8)
    asyncLoadLdsB64(ldsOffOf(&As[r * LD]) + lane * 8,
                    A + (size_t)(rowBase + r) * 128 + lane * 4);
  waitAsync0();
  __syncthreads();

  const int hs = lane >> 4, ln = lane & 15;
  const int n0 = wave * 16;
  v8f acc = {};
#pragma unroll
  for (int kc = 0; kc < 128; kc += 32) {
    v16h af, bf;
#pragma unroll
    for (int i = 0; i < 8; ++i) {
      // VGPR i, halves p=0,1 -> K = kc + (i>=4?16:0) + hs*8 + 2*(i&3) + p
      int kb = kc + ((i & 4) << 2) + hs * 8 + ((i & 3) << 1);
      af[2 * i]     = As[ln * LD + kb];          // 8-half contiguous runs -> ds_load_b128
      af[2 * i + 1] = As[ln * LD + kb + 1];
      bf[2 * i]     = Bs[(n0 + ln) * LD + kb];
      bf[2 * i + 1] = Bs[(n0 + ln) * LD + kb + 1];
    }
    acc = __builtin_amdgcn_wmma_f32_16x16x32_f16(false, af, false, bf,
                                                 (short)0, acc, false, false);
  }
#pragma unroll
  for (int r = 0; r < 8; ++r) {
    int m = rowBase + hs * 8 + r;                       // C/D: VGPR r -> M = r + hs*8
    C[(size_t)m * ldc + colBase + n0 + ln] = acc[r] * scale;
  }
}

// ---------- per-node attention scalars: sl = hw . al, sr = hw . ar ----------
__global__ void k_node_dots(const float* __restrict__ hw, const float* __restrict__ al,
                            const float* __restrict__ ar, float* __restrict__ sl,
                            float* __restrict__ sr) {
  int node = blockIdx.x * 8 + (threadIdx.x >> 5);
  int lane = threadIdx.x & 31;
  if (node >= NNODE) return;
  float4 v = ((const float4*)(hw + (size_t)node * KDIM))[lane];
  float4 a = ((const float4*)al)[lane];
  float4 b = ((const float4*)ar)[lane];
  float dl = v.x * a.x + v.y * a.y + v.z * a.z + v.w * a.w;
  float dr = v.x * b.x + v.y * b.y + v.z * b.z + v.w * b.w;
  for (int off = 16; off; off >>= 1) {
    dl += __shfl_xor(dl, off, 32);
    dr += __shfl_xor(dr, off, 32);
  }
  if (lane == 0) { sl[node] = dl; sr[node] = dr; }
}

// ---------- zero accumulators for a layer ----------
__global__ void k_init_layer(float* __restrict__ hacc, unsigned* __restrict__ menc,
                             float* __restrict__ denom) {
  size_t i = (size_t)blockIdx.x * blockDim.x + threadIdx.x;
  const size_t NK = (size_t)NNODE * KDIM;
  if (i < NK) hacc[i] = 0.f;
  if (i < NNODE) { menc[i] = 0u; denom[i] = 0.f; }
}

// ---------- edge pass 1: e = leaky_relu(sl[src]+sr[dst]); segment max ----------
__global__ void k_edge_logits(const int* __restrict__ ei, const float* __restrict__ sl,
                              const float* __restrict__ sr, unsigned* __restrict__ menc,
                              float* __restrict__ ebuf) {
  int i = blockIdx.x * 256 + threadIdx.x;
  if (i >= EEDGE) return;
  int s = ei[i], d = ei[EEDGE + i];
  float e = sl[s] + sr[d];
  e = (e > 0.f) ? e : 0.2f * e;
  ebuf[i] = e;
  atomicMax(menc + d, fenc(e));
}

// ---------- decode segment max in place (empty segment -> 0, like isfinite mask) ----------
__global__ void k_decode_max(unsigned* __restrict__ menc) {
  int n = blockIdx.x * 256 + threadIdx.x;
  if (n >= NNODE) return;
  unsigned u = menc[n];
  float m = (u == 0u) ? 0.f : fdec(u);
  reinterpret_cast<float*>(menc)[n] = m;
}

// ---------- edge pass 2: ex = exp(e - m[dst]); segment sum ----------
__global__ void k_edge_exp(const int* __restrict__ ei, float* __restrict__ ebuf,
                           const float* __restrict__ m, float* __restrict__ denom) {
  int i = blockIdx.x * 256 + threadIdx.x;
  if (i >= EEDGE) return;
  int d = ei[EEDGE + i];
  float v = expf(ebuf[i] - m[d]);
  ebuf[i] = v;
  atomAddF(denom + d, v);
}

// ---------- edge pass 3: hacc[dst] += alpha * hw[src]  (one wave per edge) ----------
__global__ void k_edge_aggr(const int* __restrict__ ei, const float* __restrict__ ebuf,
                            const float* __restrict__ denom, const float* __restrict__ hw,
                            float* __restrict__ hacc) {
  int e = blockIdx.x * 8 + (threadIdx.x >> 5);
  int lane = threadIdx.x & 31;
  if (e >= EEDGE) return;
  int s = ei[e], d = ei[EEDGE + e];
  float alpha = ebuf[e] / (denom[d] + 1e-9f);
  float4 v = ((const float4*)(hw + (size_t)s * KDIM))[lane];
  float* o = hacc + (size_t)d * KDIM + lane * 4;
  atomAddF(o + 0, alpha * v.x);
  atomAddF(o + 1, alpha * v.y);
  atomAddF(o + 2, alpha * v.z);
  atomAddF(o + 3, alpha * v.w);
}

// ---------- elu (f32 + f16 shadow) ----------
__global__ void k_elu(const float* __restrict__ hacc, float* __restrict__ h,
                      _Float16* __restrict__ h16) {
  size_t i = (size_t)blockIdx.x * blockDim.x + threadIdx.x;
  if (i >= (size_t)NNODE * KDIM) return;
  float x = hacc[i];
  float y = (x > 0.f) ? x : (expf(x) - 1.f);
  h[i] = y;
  h16[i] = (_Float16)y;
}

// ---------- prediction head ----------
__global__ __launch_bounds__(128)
void k_predict(const float* __restrict__ h, const int* __restrict__ stu_ids,
               const int* __restrict__ exer_ids, const float* __restrict__ Wf1,
               const float* __restrict__ Wf2, const float* __restrict__ Wf3,
               const float* __restrict__ bf3, float* __restrict__ out) {
  __shared__ float bs[KDIM], be[KDIM], red[KDIM];
  int b = blockIdx.x, k = threadIdx.x;
  bs[k] = h[(size_t)stu_ids[b] * KDIM + k];
  be[k] = h[(size_t)(N_STU + exer_ids[b]) * KDIM + k];
  __syncthreads();
  float p = 0.f, dif = 0.f;
#pragma unroll 4
  for (int t = 0; t < KDIM; ++t) {
    p   += bs[t] * Wf1[k * KDIM + t];
    dif += be[t] * Wf2[k * KDIM + t];
  }
  red[k] = Wf3[k] * (sigmoidf(p) - sigmoidf(dif));
  __syncthreads();
  for (int s = 64; s; s >>= 1) { if (k < s) red[k] += red[k + s]; __syncthreads(); }
  if (k == 0) out[b] = sigmoidf(red[0] + bf3[0]);
}

// ---------- gather contrastive views ----------
__global__ void k_gather_views(const float* __restrict__ h, const int* __restrict__ vs,
                               const int* __restrict__ ve, float* __restrict__ zs,
                               float* __restrict__ ze) {
  int i = blockIdx.x, k = threadIdx.x;
  zs[(size_t)i * KDIM + k] = h[(size_t)vs[i] * KDIM + k];
  ze[(size_t)i * KDIM + k] = h[(size_t)(N_STU + ve[i]) * KDIM + k];
}

// ---------- row L2 normalize: f32 in -> f16 out, grid = #rows ----------
__global__ __launch_bounds__(128)
void k_l2norm(const float* __restrict__ z, _Float16* __restrict__ z16) {
  __shared__ float red[128];
  int i = blockIdx.x, k = threadIdx.x;
  float x = z[(size_t)i * KDIM + k];
  red[k] = x * x;
  __syncthreads();
  for (int s = 64; s; s >>= 1) { if (k < s) red[k] += red[k + s]; __syncthreads(); }
  float nrm = fmaxf(sqrtf(red[0]), 1e-8f);
  z16[(size_t)i * KDIM + k] = (_Float16)(x / nrm);
}

__global__ void k_zero(float* __restrict__ p, int n) {
  int i = blockIdx.x * blockDim.x + threadIdx.x;
  if (i < n) p[i] = 0.f;
}

// ---------- NT-Xent reductions: acc += log(sum_j exp(sim_ij)) - sim_ii ----------
__global__ __launch_bounds__(256)
void k_lse_row(const float* __restrict__ sim, float* __restrict__ acc) {
  __shared__ float red[256];
  int i = blockIdx.x, t = threadIdx.x;
  const float* row = sim + (size_t)i * BCV;
  float s = 0.f;
  for (int j = t; j < BCV; j += 256) s += expf(row[j]);
  red[t] = s; __syncthreads();
  for (int st = 128; st; st >>= 1) { if (t < st) red[t] += red[t + st]; __syncthreads(); }
  if (t == 0) atomAddF(acc, logf(red[0]) - row[i]);
}

__global__ __launch_bounds__(256)
void k_lse_col(const float* __restrict__ sim, float* __restrict__ acc) {
  __shared__ float red[256];
  int i = blockIdx.x, t = threadIdx.x;
  float s = 0.f;
  for (int j = t; j < BCV; j += 256) s += expf(sim[(size_t)j * BCV + i]);
  red[t] = s; __syncthreads();
  for (int st = 128; st; st >>= 1) { if (t < st) red[t] += red[t + st]; __syncthreads(); }
  if (t == 0) atomAddF(acc, logf(red[0]) - sim[(size_t)i * BCV + i]);
}

__global__ void k_finalize(const float* __restrict__ acc, float* __restrict__ out) {
  out[BB] = 0.5f * acc[0] / (float)BCV;
}

// ====================================================================
extern "C" void kernel_launch(void* const* d_in, const int* in_sizes, int n_in,
                              void* d_out, int out_size, void* d_ws, size_t ws_size,
                              hipStream_t stream) {
  (void)in_sizes; (void)n_in; (void)out_size; (void)ws_size;
  const float* stu  = (const float*)d_in[0];
  const float* exer = (const float*)d_in[1];
  const float* ktab = (const float*)d_in[2];
  const float* W1   = (const float*)d_in[3];
  const float* a1l  = (const float*)d_in[4];
  const float* a1r  = (const float*)d_in[5];
  const float* W2   = (const float*)d_in[6];
  const float* a2l  = (const float*)d_in[7];
  const float* a2r  = (const float*)d_in[8];
  const float* Wf1  = (const float*)d_in[9];
  const float* Wf2  = (const float*)d_in[10];
  const float* Wf3  = (const float*)d_in[11];
  const float* bf3  = (const float*)d_in[12];
  // d_in[13] = kn_r: unused — sum(o*kn_r)/sum(kn_r) == o since o is (B,1) broadcast
  const int* stu_ids  = (const int*)d_in[14];
  const int* exer_ids = (const int*)d_in[15];
  const int* ei   = (const int*)d_in[16];
  const int* eig1 = (const int*)d_in[17];
  const int* eig2 = (const int*)d_in[18];
  const int* vstu = (const int*)d_in[19];
  const int* vexer = (const int*)d_in[20];
  float* out = (float*)d_out;

  // ---- workspace carve (float units) ----
  float* ws = (float*)d_ws;
  const size_t NK = (size_t)NNODE * KDIM;
  size_t off = 0;
  float* hA   = ws + off; off += NK;        // node features f32 (layer in/out)
  float* hW   = ws + off; off += NK;        // hw = h @ W
  float* hACC = ws + off; off += NK;        // aggregation accumulator
  float* sl   = ws + off; off += NNODE;
  float* sr   = ws + off; off += NNODE;
  unsigned* menc = (unsigned*)(ws + off); off += NNODE;  // seg-max (uint), then float m
  float* denom = ws + off; off += NNODE;
  float* ebuf  = ws + off; off += EEDGE;    // per-edge e, then exp(e-m)
  float* zs1 = ws + off; off += (size_t)BCV * KDIM;
  float* zs2 = ws + off; off += (size_t)BCV * KDIM;
  float* ze1 = ws + off; off += (size_t)BCV * KDIM;
  float* ze2 = ws + off; off += (size_t)BCV * KDIM;
  float* acc = ws + off; off += 4;
  _Float16* W1t = (_Float16*)(ws + off); off += (KDIM * KDIM) / 2;   // W1^T, f16
  _Float16* W2t = (_Float16*)(ws + off); off += (KDIM * KDIM) / 2;   // W2^T, f16
  _Float16* hAh = (_Float16*)(ws + off); off += NK / 2;              // f16 shadow of hA
  _Float16* z16 = (_Float16*)(ws + off); off += (4 * (size_t)BCV * KDIM) / 2;
  float* sim = ws;   // 2048^2 f32 aliases hA/hW head: both dead when sims run

  _Float16* zs1h = z16;
  _Float16* zs2h = z16 + (size_t)BCV * KDIM;
  _Float16* ze1h = z16 + 2 * (size_t)BCV * KDIM;
  _Float16* ze2h = z16 + 3 * (size_t)BCV * KDIM;

  const dim3 B256(256);
  const unsigned gNK = (unsigned)((NK + 255) / 256);

  // one-time weight transpose + conversion to f16
  k_transpose_f16<<<dim3((KDIM * KDIM + 255) / 256), B256, 0, stream>>>(W1, W1t);
  k_transpose_f16<<<dim3((KDIM * KDIM + 255) / 256), B256, 0, stream>>>(W2, W2t);

  auto layer = [&](const int* edges, const _Float16* Wt, const float* al, const float* ar) {
    k_gemm128<<<dim3(NNODE / 16, 1), B256, 0, stream>>>(hAh, Wt, hW, KDIM, 1.0f);
    k_node_dots<<<dim3((NNODE + 7) / 8), B256, 0, stream>>>(hW, al, ar, sl, sr);
    k_init_layer<<<dim3(gNK), B256, 0, stream>>>(hACC, menc, denom);
    k_edge_logits<<<dim3((EEDGE + 255) / 256), B256, 0, stream>>>(edges, sl, sr, menc, ebuf);
    k_decode_max<<<dim3((NNODE + 255) / 256), B256, 0, stream>>>(menc);
    k_edge_exp<<<dim3((EEDGE + 255) / 256), B256, 0, stream>>>(edges, ebuf,
                                                              (const float*)menc, denom);
    k_edge_aggr<<<dim3((EEDGE + 7) / 8), B256, 0, stream>>>(edges, ebuf, denom, hW, hACC);
    k_elu<<<dim3(gNK), B256, 0, stream>>>(hACC, hA, hAh);
  };
  auto encode = [&](const int* edges) {
    k_build_h0<<<dim3(gNK), B256, 0, stream>>>(hA, hAh, stu, exer, ktab);
    layer(edges, W1t, a1l, a1r);
    layer(edges, W2t, a2l, a2r);
  };

  // main graph -> prediction output
  encode(ei);
  k_predict<<<dim3(BB), dim3(128), 0, stream>>>(hA, stu_ids, exer_ids, Wf1, Wf2, Wf3, bf3, out);

  // contrastive views
  encode(eig1);
  k_gather_views<<<dim3(BCV), dim3(128), 0, stream>>>(hA, vstu, vexer, zs1, ze1);
  encode(eig2);
  k_gather_views<<<dim3(BCV), dim3(128), 0, stream>>>(hA, vstu, vexer, zs2, ze2);

  k_l2norm<<<dim3(4 * BCV), dim3(128), 0, stream>>>(zs1, z16);  // zs1..ze2 contiguous
  k_zero<<<dim3(1), dim3(32), 0, stream>>>(acc, 4);

  // student pair: sim = zs1 zs2^T / T ; nt_xent both directions = row-LSE + col-LSE
  k_gemm128<<<dim3(BCV / 16, BCV / 128), B256, 0, stream>>>(zs1h, zs2h, sim, BCV, INV_TEMP);
  k_lse_row<<<dim3(BCV), B256, 0, stream>>>(sim, acc);
  k_lse_col<<<dim3(BCV), B256, 0, stream>>>(sim, acc);
  // exercise pair
  k_gemm128<<<dim3(BCV / 16, BCV / 128), B256, 0, stream>>>(ze1h, ze2h, sim, BCV, INV_TEMP);
  k_lse_row<<<dim3(BCV), B256, 0, stream>>>(sim, acc);
  k_lse_col<<<dim3(BCV), B256, 0, stream>>>(sim, acc);

  k_finalize<<<dim3(1), dim3(1), 0, stream>>>(acc, out);
}
